// LearnableChannelSplit_6957847019658
// MI455X (gfx1250) — matrix operand order repaired and checked by reading, not compiled, not run on hardware
//
#include <hip/hip_runtime.h>
#include <hip/hip_bf16.h>

#define B_IMG 16
#define C_IN 512
#define HW 9216                       // 96*96
#define CH 128
#define NPIX (B_IMG * HW)             // 147456
#define PIX_PER_WAVE 16
#define WAVES_PER_BLOCK 8
#define PIX_PER_BLOCK (PIX_PER_WAVE * WAVES_PER_BLOCK)   // 128
#define W1_LDS_STRIDE 520             // 512 + 8 bf16 pad (16B) -> kills LDS bank conflicts

typedef __attribute__((ext_vector_type(16))) __bf16 v16bf;
typedef __attribute__((ext_vector_type(8)))  __bf16 v8bf;
typedef __attribute__((ext_vector_type(8)))  float  v8f;

__launch_bounds__(256)
__global__ void mask_split_kernel(const float* __restrict__ X,
                                  const float* __restrict__ W1,
                                  const float* __restrict__ B1,
                                  const float* __restrict__ W2,
                                  const float* __restrict__ B2,
                                  float* __restrict__ OUT) {
  extern __shared__ __align__(16) unsigned char smem[];
  __bf16* w1s = (__bf16*)smem;                                    // 128*520 bf16 = 133120 B
  float*  b1s = (float*)(smem + CH * W1_LDS_STRIDE * 2);          // 128 f32
  float2* w2s = (float2*)(smem + CH * W1_LDS_STRIDE * 2 + CH * 4);// 128 float2 {w2[0][o], w2[1][o]}

  const int tid  = threadIdx.x;
  const int lane = tid & 31;
  const int wid  = tid >> 5;
  const int col  = lane & 15;   // pixel within wave tile / matrix column
  const int hi   = lane >> 4;   // half-wave select

  // ---- stage w1 (fp32 -> bf16), b1, w2 into LDS ----
  for (int idx = tid; idx < CH * (C_IN / 4); idx += 256) {
    int o  = idx >> 7;            // row (out channel)
    int c4 = (idx & 127) << 2;    // 4-wide column group
    float4 w = *(const float4*)(W1 + o * C_IN + c4);
    __bf16* dst = w1s + o * W1_LDS_STRIDE + c4;
    dst[0] = (__bf16)w.x; dst[1] = (__bf16)w.y;
    dst[2] = (__bf16)w.z; dst[3] = (__bf16)w.w;
  }
  if (tid < CH) {
    b1s[tid] = B1[tid];
    w2s[tid] = make_float2(W2[tid], W2[CH + tid]);
  }
  __syncthreads();

  // ---- tile base (9216 % 128 == 0 -> a block never straddles an image) ----
  const int p0   = blockIdx.x * PIX_PER_BLOCK + wid * PIX_PER_WAVE;
  const int bimg = p0 / HW;
  const int hw0  = p0 - bimg * HW;

  const float* xbase = X + (size_t)bimg * C_IN * HW + hw0 + col;

  v8f acc[8];
  #pragma unroll
  for (int j = 0; j < 8; ++j) acc[j] = (v8f)0.0f;

  // ---- GEMM1: h[o][pix] = sum_c w1[o][c] * x[c][pix], K=512 in steps of 32 ----
  for (int k0 = 0; k0 < C_IN; k0 += 32) {
    // B fragment (x): VGPR r holds K = 2r,2r+1 (+16 for upper half-wave), N = col
    v16bf bfrag;
    const float* xk = xbase + (size_t)(k0 + 16 * hi) * HW;
    #pragma unroll
    for (int r = 0; r < 8; ++r) {
      float f0 = xk[(size_t)(2 * r) * HW];
      float f1 = xk[(size_t)(2 * r + 1) * HW];
      bfrag[2 * r]     = (__bf16)f0;
      bfrag[2 * r + 1] = (__bf16)f1;
    }
    #pragma unroll
    for (int j = 0; j < 8; ++j) {
      // A fragment (w1): lane row M = 16j+col; K = k0 + 16*hi + {0..15}
      const __bf16* arow = w1s + (16 * j + col) * W1_LDS_STRIDE + k0 + 16 * hi;
      v8bf alo  = *(const v8bf*)arow;          // ds_load_b128
      v8bf ahi2 = *(const v8bf*)(arow + 8);    // ds_load_b128
      v16bf afrag = __builtin_shufflevector(alo, ahi2,
          0, 1, 2, 3, 4, 5, 6, 7, 8, 9, 10, 11, 12, 13, 14, 15);
      acc[j] = __builtin_amdgcn_wmma_f32_16x16x32_bf16(
          false, afrag, false, bfrag, (short)0, acc[j], false, false);
    }
  }

  // ---- GEMM2 (128 -> 2) + bias + ReLU folded in, then 2-way softmax ----
  float l0 = 0.f, l1 = 0.f;
  #pragma unroll
  for (int j = 0; j < 8; ++j) {
    #pragma unroll
    for (int r = 0; r < 8; ++r) {
      int o = 16 * j + r + 8 * hi;        // C layout: VGPR r -> M = r + 8*hi
      float hval = acc[j][r] + b1s[o];
      hval = fmaxf(hval, 0.f);
      float2 w = w2s[o];
      l0 = fmaf(w.x, hval, l0);
      l1 = fmaf(w.y, hval, l1);
    }
  }
  l0 += __shfl_xor(l0, 16, 32);           // combine the two half-wave partials
  l1 += __shfl_xor(l1, 16, 32);
  l0 += B2[0];
  l1 += B2[1];

  float m0 = 1.0f / (1.0f + __expf(l1 - l0));   // softmax over 2 channels
  float m1 = 1.0f / (1.0f + __expf(l0 - l1));

  float* X0 = OUT;
  float* X1 = OUT + (size_t)B_IMG * C_IN * HW;
  float* M  = OUT + (size_t)2 * B_IMG * C_IN * HW;

  // m: lanes 0-15 write channel 0, lanes 16-31 write channel 1 (coalesced)
  M[((size_t)bimg * 2 + hi) * HW + hw0 + col] = hi ? m1 : m0;

  // ---- split phase: x0 = x*m0, x1 = x*m1 (float4 per lane, 8 channel rows/iter) ----
  int pq = (lane & 3) << 2;               // 4-pixel group this lane handles
  float m0v[4], m1v[4];
  #pragma unroll
  for (int i = 0; i < 4; ++i) {
    m0v[i] = __shfl(m0, pq + i, 32);
    m1v[i] = __shfl(m1, pq + i, 32);
  }
  const size_t base = (size_t)bimg * C_IN * HW + hw0 + pq;
  for (int c = lane >> 2; c < C_IN; c += 8) {
    size_t off = base + (size_t)c * HW;
    float4 v = *(const float4*)(X + off);
    float4 a, b;
    a.x = v.x * m0v[0]; a.y = v.y * m0v[1]; a.z = v.z * m0v[2]; a.w = v.w * m0v[3];
    b.x = v.x * m1v[0]; b.y = v.y * m1v[1]; b.z = v.z * m1v[2]; b.w = v.w * m1v[3];
    *(float4*)(X0 + off) = a;
    *(float4*)(X1 + off) = b;
  }
}

extern "C" void kernel_launch(void* const* d_in, const int* in_sizes, int n_in,
                              void* d_out, int out_size, void* d_ws, size_t ws_size,
                              hipStream_t stream) {
  (void)in_sizes; (void)n_in; (void)out_size; (void)d_ws; (void)ws_size;
  const float* X  = (const float*)d_in[0];
  const float* W1 = (const float*)d_in[1];
  const float* B1 = (const float*)d_in[2];
  const float* W2 = (const float*)d_in[3];
  const float* B2 = (const float*)d_in[4];
  float* OUT = (float*)d_out;

  const int nblocks = NPIX / PIX_PER_BLOCK;                       // 1152
  const size_t smem = (size_t)CH * W1_LDS_STRIDE * 2              // w1 bf16
                    + (size_t)CH * 4                              // b1
                    + (size_t)CH * 8;                             // w2 pairs
  mask_split_kernel<<<nblocks, 256, smem, stream>>>(X, W1, B1, W2, B2, OUT);
}
